// EdgePredictor_35493609734344
// MI455X (gfx1250) — compile-verified, matrix operand order checked by reading, not run on hardware
//
#include <hip/hip_runtime.h>
#include <math.h>

typedef __attribute__((ext_vector_type(16))) _Float16 v16h;
typedef __attribute__((ext_vector_type(8)))  _Float16 v8h;
typedef __attribute__((ext_vector_type(4)))  _Float16 v4h;
typedef __attribute__((ext_vector_type(8)))  float    v8f;

#define D_MEM 1024
#define D_HID 256
#define KS1   72   // 2304/32
#define KS2   8    // 256/32
#define KS0   2    // 64/32 (spatial encoder layer 2)
#define MT    4    // 16-row M sub-tiles per block
#define MROWS 64   // pairs per block

#define STRD  264                 // f16 tile row stride (528B: 16B aligned, bank-spread)
#define R0_BYTES (MROWS * STRD * 2)   // 33792: se tile, then h2
#define R1_BYTES (MROWS * STRD * 2)   // 33792: sp+hid, then LN stats, then h3
#define SMEM_BYTES (R0_BYTES + R1_BYTES) // 67584 (dynamic LDS; WGP has 320KB)

// ---------------- WMMA helpers ----------------

__device__ __forceinline__ v8f wmma_f16(v16h a, v16h b, v8f c) {
  return __builtin_amdgcn_wmma_f32_16x16x32_f16(false, a, false, b, (short)0, c,
                                                false, false);
}

// B fragment from packed weights: layout [ntile][ks][lane][16 halves]
__device__ __forceinline__ v16h b_frag(const _Float16* __restrict__ base,
                                       int ntile, int ks, int lane, int ksteps) {
  const v8h* p = (const v8h*)(base + (((size_t)(ntile * ksteps + ks) * 32 + lane) << 4));
  union { v16h v; v8h h[2]; } u;
  u.h[0] = p[0];
  u.h[1] = p[1];
  return u.v;
}

// A fragment from an f16 row (global f16 table or LDS tile).
// 16-bit A layout: lanes 0-15 hold K chunks [0,8)+[16,24); lanes 16-31 hold [8,16)+[24,32).
__device__ __forceinline__ v16h a_frag_h(const _Float16* row, int kbase, int laneHi) {
  int c0 = kbase + (laneHi << 3);
  union { v16h v; v8h h[2]; } u;
  u.h[0] = *(const v8h*)(row + c0);
  u.h[1] = *(const v8h*)(row + c0 + 16);
  return u.v;
}

__device__ __forceinline__ float sigmoidf_(float x) { return 1.0f / (1.0f + __expf(-x)); }

// ---------------- prep kernels ----------------

// Convert the f32 gather table to f16 (41MB; L2-resident thereafter).
__global__ void cvt_table(const float* __restrict__ in, _Float16* __restrict__ out,
                          long n4) {
  long i = (long)blockIdx.x * blockDim.x + threadIdx.x;
  if (i >= n4) return;
  float4 v = ((const float4*)in)[i];
  v4h h;
  h[0] = (_Float16)v.x; h[1] = (_Float16)v.y; h[2] = (_Float16)v.z; h[3] = (_Float16)v.w;
  ((v4h*)out)[i] = h;
}

__global__ void pack_weight(const float* __restrict__ W, _Float16* __restrict__ out,
                            int ksteps, int ncols) {
  int tid = blockIdx.x * blockDim.x + threadIdx.x;
  int total = ksteps * (ncols / 16) * 32;
  if (tid >= total) return;
  int lane   = tid & 31;
  int ks     = (tid >> 5) % ksteps;
  int nt     = (tid >> 5) / ksteps;
  int laneHi = lane >> 4;
  int col    = nt * 16 + (lane & 15);
  _Float16* dst = out + ((size_t)tid << 4);
  int kbase = ks * 32;
#pragma unroll
  for (int j = 0; j < 16; ++j) {
    int k = kbase + j + 8 * ((j >> 3) + laneHi);
    dst[j] = (_Float16)W[(size_t)k * ncols + col];
  }
}

// Heads packed as 256 x 32: cols 0..2 = W_a, 3..8 = W_s, 9..25 = W_c, rest 0.
__global__ void pack_heads(const float* __restrict__ Wa, const float* __restrict__ Ws,
                           const float* __restrict__ Wc, const float* __restrict__ ba,
                           const float* __restrict__ bs, const float* __restrict__ bc,
                           _Float16* __restrict__ out, float* __restrict__ biasOut) {
  int tid = blockIdx.x * blockDim.x + threadIdx.x;
  int total = KS2 * 2 * 32;
  if (tid < total) {
    int lane   = tid & 31;
    int ks     = (tid >> 5) % KS2;
    int nt     = (tid >> 5) / KS2;
    int laneHi = lane >> 4;
    int col    = nt * 16 + (lane & 15);
    _Float16* dst = out + ((size_t)tid << 4);
    int kbase = ks * 32;
#pragma unroll
    for (int j = 0; j < 16; ++j) {
      int k = kbase + j + 8 * ((j >> 3) + laneHi);
      float v = 0.0f;
      if (col < 3)       v = Wa[(size_t)k * 3 + col];
      else if (col < 9)  v = Ws[(size_t)k * 6 + (col - 3)];
      else if (col < 26) v = Wc[(size_t)k * 17 + (col - 9)];
      dst[j] = (_Float16)v;
    }
  }
  if (tid < 32) {
    float b = 0.0f;
    if (tid < 3)       b = ba[tid];
    else if (tid < 9)  b = bs[tid - 3];
    else if (tid < 26) b = bc[tid - 9];
    biasOut[tid] = b;
  }
}

// ---------------- main fused kernel ----------------

__global__ __launch_bounds__(256)
void edge_main(const _Float16* __restrict__ tab, const int* __restrict__ pidx,
               const int* __restrict__ oidx, const float* __restrict__ corners,
               const float* __restrict__ Wse1, const float* __restrict__ bse1,
               const float* __restrict__ bse2,
               const float* __restrict__ bp1, const float* __restrict__ lng,
               const float* __restrict__ lnb, const float* __restrict__ bp2,
               const _Float16* __restrict__ Wp1p, const _Float16* __restrict__ Wse2p,
               const _Float16* __restrict__ Wp2p, const _Float16* __restrict__ Whp,
               const float* __restrict__ hbias,
               float* __restrict__ out, int K) {
  extern __shared__ char smemRaw[];
  // Region 0: SE output tile during GEMM1, then layernormed h2 (both f16 [64][STRD]).
  _Float16* seBuf = (_Float16*)smemRaw;
  _Float16* h2Buf = (_Float16*)smemRaw;
  // Region 1: phase0 sp+hid; LN stats window; then h3 (f16 [64][STRD]).
  float*    spBuf  = (float*)(smemRaw + R0_BYTES);             // [64][8]
  _Float16* hidBuf = (_Float16*)(smemRaw + R0_BYTES + 2048);   // [64][72] f16
  float*    psum   = (float*)(smemRaw + R0_BYTES);             // [64]
  float*    psq    = psum + 64;                                 // [64]
  float*    muB    = psum + 128;                                // [64]
  float*    rsB    = psum + 192;                                // [64]
  _Float16* h3Buf  = (_Float16*)(smemRaw + R0_BYTES);          // [64][STRD]

  const int t        = threadIdx.x;
  const int lane     = t & 31;
  const int w        = t >> 5;      // wave 0..7
  const int laneHi   = lane >> 4;
  const int m        = lane & 15;
  const int tileBase = blockIdx.x * MROWS;
  const int t0 = 2 * w, t1 = 2 * w + 1;  // this wave's N-tiles

  // ---- Phase 0a: spatial features, one thread per pair ----
  if (t < MROWS) {
    int gm = tileBase + t; if (gm > K - 1) gm = K - 1;
    int pi = pidx[gm], oi = oidx[gm];
    const float* pc = corners + (size_t)pi * 24;
    const float* oc = corners + (size_t)oi * 24;
    float psm[3] = {0, 0, 0}, osm[3] = {0, 0, 0};
    float pmn[3] = {1e30f, 1e30f, 1e30f}, pmx[3] = {-1e30f, -1e30f, -1e30f};
    float omn[3] = {1e30f, 1e30f, 1e30f}, omx[3] = {-1e30f, -1e30f, -1e30f};
#pragma unroll
    for (int j = 0; j < 8; ++j)
#pragma unroll
      for (int c = 0; c < 3; ++c) {
        float pv = pc[j * 3 + c], ov = oc[j * 3 + c];
        psm[c] += pv; osm[c] += ov;
        pmn[c] = fminf(pmn[c], pv); pmx[c] = fmaxf(pmx[c], pv);
        omn[c] = fminf(omn[c], ov); omx[c] = fmaxf(omx[c], ov);
      }
    float rel[3]; float pvol = 1.0f, ovol = 1.0f;
#pragma unroll
    for (int c = 0; c < 3; ++c) {
      rel[c] = osm[c] * 0.125f - psm[c] * 0.125f;
      pvol *= fmaxf(pmx[c] - pmn[c], 1e-6f);
      ovol *= fmaxf(omx[c] - omn[c], 1e-6f);
    }
    float dist = sqrtf(rel[0] * rel[0] + rel[1] * rel[1] + rel[2] * rel[2]);
    float lvr  = logf(ovol + 1e-6f) - logf(pvol + 1e-6f);
    spBuf[t * 8 + 0] = rel[0]; spBuf[t * 8 + 1] = rel[1]; spBuf[t * 8 + 2] = rel[2];
    spBuf[t * 8 + 3] = dist;   spBuf[t * 8 + 4] = lvr;
  }
  __syncthreads();

  // ---- Phase 0b: SE hidden (relu(sp @ Wse1 + b)) -> hid f16 [64][72] ----
  {
    int p = t >> 2;                       // 4 threads per pair
#pragma unroll
    for (int i = 0; i < 16; ++i) {
      int o = (t & 3) * 16 + i;
      float acc = bse1[o];
#pragma unroll
      for (int k = 0; k < 5; ++k) acc += spBuf[p * 8 + k] * Wse1[k * 64 + o];
      hidBuf[p * 72 + o] = (_Float16)fmaxf(acc, 0.0f);
    }
  }
  __syncthreads();

  // ---- Phase 0c: SE layer 2 via WMMA: hid[64,64] @ Wse2 -> se tile (+bias) ----
  {
    float bb0 = bse2[w * 32 + m], bb1 = bse2[w * 32 + m + 16];
#pragma unroll
    for (int s = 0; s < MT; ++s) {
      v8f a0 = {}; v8f a1 = {};
      const _Float16* hrow = hidBuf + (size_t)(s * 16 + m) * 72;
#pragma unroll
      for (int ks = 0; ks < KS0; ++ks) {
        v16h a  = a_frag_h(hrow, ks * 32, laneHi);
        v16h b0 = b_frag(Wse2p, t0, ks, lane, KS0);
        v16h b1 = b_frag(Wse2p, t1, ks, lane, KS0);
        a0 = wmma_f16(a, b0, a0);
        a1 = wmma_f16(a, b1, a1);
      }
#pragma unroll
      for (int r = 0; r < 8; ++r) {
        int row = s * 16 + r + (laneHi << 3);
        seBuf[row * STRD + w * 32 + m]      = (_Float16)(a0[r] + bb0);
        seBuf[row * STRD + w * 32 + m + 16] = (_Float16)(a1[r] + bb1);
      }
    }
  }
  __syncthreads();

  // ---- Phase 1: GEMM1: [64,2304] @ Wp1 ----
  const _Float16* prow[MT];
  const _Float16* orow[MT];
#pragma unroll
  for (int s = 0; s < MT; ++s) {
    int gm = tileBase + s * 16 + m; if (gm > K - 1) gm = K - 1;
    prow[s] = tab + (size_t)pidx[gm] * D_MEM;
    orow[s] = tab + (size_t)oidx[gm] * D_MEM;
  }

  v8f acc0[MT], acc1[MT];
#pragma unroll
  for (int s = 0; s < MT; ++s) { acc0[s] = (v8f){}; acc1[s] = (v8f){}; }

  for (int ks = 0; ks < 32; ++ks) {            // person rows, K [0,1024)
    v16h b0 = b_frag(Wp1p, t0, ks, lane, KS1);
    v16h b1 = b_frag(Wp1p, t1, ks, lane, KS1);
#pragma unroll
    for (int s = 0; s < MT; ++s) {
      v16h a = a_frag_h(prow[s], ks * 32, laneHi);
      acc0[s] = wmma_f16(a, b0, acc0[s]);
      acc1[s] = wmma_f16(a, b1, acc1[s]);
    }
  }
  for (int ks = 32; ks < 64; ++ks) {           // object rows, K [1024,2048)
    v16h b0 = b_frag(Wp1p, t0, ks, lane, KS1);
    v16h b1 = b_frag(Wp1p, t1, ks, lane, KS1);
#pragma unroll
    for (int s = 0; s < MT; ++s) {
      v16h a = a_frag_h(orow[s], (ks - 32) * 32, laneHi);
      acc0[s] = wmma_f16(a, b0, acc0[s]);
      acc1[s] = wmma_f16(a, b1, acc1[s]);
    }
  }
#pragma unroll
  for (int ks = 64; ks < 72; ++ks) {           // sp_enc tile, K [2048,2304)
    v16h b0 = b_frag(Wp1p, t0, ks, lane, KS1);
    v16h b1 = b_frag(Wp1p, t1, ks, lane, KS1);
#pragma unroll
    for (int s = 0; s < MT; ++s) {
      v16h a = a_frag_h(seBuf + (size_t)(s * 16 + m) * STRD, (ks - 64) * 32, laneHi);
      acc0[s] = wmma_f16(a, b0, acc0[s]);
      acc1[s] = wmma_f16(a, b1, acc1[s]);
    }
  }

  // bias + relu in registers
  {
    float bb0 = bp1[w * 32 + m], bb1 = bp1[w * 32 + m + 16];
#pragma unroll
    for (int s = 0; s < MT; ++s)
#pragma unroll
      for (int r = 0; r < 8; ++r) {
        acc0[s][r] = fmaxf(acc0[s][r] + bb0, 0.0f);
        acc1[s][r] = fmaxf(acc1[s][r] + bb1, 0.0f);
      }
  }
  __syncthreads();          // seBuf dead; R1 hid dead

  // ---- Phase 2: LayerNorm stats via LDS float atomics ----
  if (t < 128) { psum[t & 63] = 0.0f; psq[t & 63] = 0.0f; }  // init psum[64]+psq[64]
  __syncthreads();
#pragma unroll
  for (int s = 0; s < MT; ++s)
#pragma unroll
    for (int r = 0; r < 8; ++r) {
      int row = s * 16 + r + (laneHi << 3);
      float v0 = acc0[s][r], v1 = acc1[s][r];
      atomicAdd(&psum[row], v0 + v1);
      atomicAdd(&psq[row], v0 * v0 + v1 * v1);
    }
  __syncthreads();
  if (t < MROWS) {
    float mu  = psum[t] * (1.0f / 256.0f);
    float var = psq[t] * (1.0f / 256.0f) - mu * mu;
    muB[t] = mu;
    rsB[t] = rsqrtf(var + 1e-5f);
  }
  __syncthreads();

  // normalize in registers, write h2 (f16) to region 0
  {
    int n0 = w * 32 + m, n1 = n0 + 16;
    float g0 = lng[n0], g1 = lng[n1], bb0 = lnb[n0], bb1 = lnb[n1];
#pragma unroll
    for (int s = 0; s < MT; ++s)
#pragma unroll
      for (int r = 0; r < 8; ++r) {
        int row = s * 16 + r + (laneHi << 3);
        float mu = muB[row], rs = rsB[row];
        h2Buf[row * STRD + n0] = (_Float16)((acc0[s][r] - mu) * rs * g0 + bb0);
        h2Buf[row * STRD + n1] = (_Float16)((acc1[s][r] - mu) * rs * g1 + bb1);
      }
  }
  __syncthreads();

  // ---- Phase 3: GEMM2: h2[64,256] @ Wp2, bias+relu -> h3 (region 1) ----
  {
    v8f c0[MT], c1[MT];
#pragma unroll
    for (int s = 0; s < MT; ++s) { c0[s] = (v8f){}; c1[s] = (v8f){}; }
#pragma unroll
    for (int ks = 0; ks < KS2; ++ks) {
      v16h b0 = b_frag(Wp2p, t0, ks, lane, KS2);
      v16h b1 = b_frag(Wp2p, t1, ks, lane, KS2);
#pragma unroll
      for (int s = 0; s < MT; ++s) {
        v16h a = a_frag_h(h2Buf + (size_t)(s * 16 + m) * STRD, ks * 32, laneHi);
        c0[s] = wmma_f16(a, b0, c0[s]);
        c1[s] = wmma_f16(a, b1, c1[s]);
      }
    }
    float bb0 = bp2[w * 32 + m], bb1 = bp2[w * 32 + m + 16];
    __syncthreads();   // LN stats window (region 1) no longer needed
#pragma unroll
    for (int s = 0; s < MT; ++s)
#pragma unroll
      for (int r = 0; r < 8; ++r) {
        int row = s * 16 + r + (laneHi << 3);
        h3Buf[row * STRD + w * 32 + m]      = (_Float16)fmaxf(c0[s][r] + bb0, 0.0f);
        h3Buf[row * STRD + w * 32 + m + 16] = (_Float16)fmaxf(c1[s][r] + bb1, 0.0f);
      }
  }
  __syncthreads();

  // ---- Phase 4: heads (256x32 padded GEMM; wave w -> subtile w>>1, ntile w&1) ----
  {
    int s  = w >> 1;
    int nt = w & 1;
    v8f acc = {};
    const _Float16* h3row = h3Buf + (size_t)(s * 16 + m) * STRD;
#pragma unroll
    for (int ks = 0; ks < KS2; ++ks) {
      v16h a = a_frag_h(h3row, ks * 32, laneHi);
      v16h b = b_frag(Whp, nt, ks, lane, KS2);
      acc = wmma_f16(a, b, acc);
    }
    int n = nt * 16 + m;          // 0..31; 26 valid
    float bh = hbias[n];
    const size_t K3 = (size_t)3 * K, K9 = (size_t)9 * K;
#pragma unroll
    for (int r = 0; r < 8; ++r) {
      int gm = tileBase + s * 16 + r + (laneHi << 3);
      if (gm < K) {
        float v = acc[r] + bh;
        if (n < 3)       out[(size_t)gm * 3 + n] = v;                        // att logits
        else if (n < 9)  out[K3 + (size_t)gm * 6 + (n - 3)] = sigmoidf_(v);  // sp_dist
        else if (n < 26) out[K9 + (size_t)gm * 17 + (n - 9)] = sigmoidf_(v); // ct_dist
      }
    }
  }
}

// ---------------- launcher ----------------

extern "C" void kernel_launch(void* const* d_in, const int* in_sizes, int n_in,
                              void* d_out, int out_size, void* d_ws, size_t ws_size,
                              hipStream_t stream) {
  const float* es      = (const float*)d_in[0];
  const int*   pidx    = (const int*)d_in[1];
  const int*   oidx    = (const int*)d_in[2];
  const float* corners = (const float*)d_in[3];
  const float* Wse1    = (const float*)d_in[4];
  const float* bse1    = (const float*)d_in[5];
  const float* Wse2    = (const float*)d_in[6];
  const float* bse2    = (const float*)d_in[7];
  const float* Wp1     = (const float*)d_in[8];
  const float* bp1     = (const float*)d_in[9];
  const float* lng     = (const float*)d_in[10];
  const float* lnb     = (const float*)d_in[11];
  const float* Wp2     = (const float*)d_in[12];
  const float* bp2     = (const float*)d_in[13];
  const float* Wa      = (const float*)d_in[14];
  const float* ba      = (const float*)d_in[15];
  const float* Ws      = (const float*)d_in[16];
  const float* bs      = (const float*)d_in[17];
  const float* Wc      = (const float*)d_in[18];
  const float* bc      = (const float*)d_in[19];
  float* out = (float*)d_out;
  const int K = in_sizes[1];
  const long nES = in_sizes[0];          // N_NODES * D_MEM

  // workspace: f16 table | Wp1 packed | Wse2 packed | Wp2 packed | heads packed | heads bias
  _Float16* tab   = (_Float16*)d_ws;
  _Float16* wp1p  = tab + nES;
  _Float16* wse2p = wp1p + (size_t)KS1 * 16 * 32 * 16;     // 589824 halves
  _Float16* wp2p  = wse2p + (size_t)KS0 * 16 * 32 * 16;    // 16384 halves
  _Float16* whp   = wp2p + (size_t)KS2 * 16 * 32 * 16;     // 65536 halves
  float*    hb    = (float*)(whp + (size_t)KS2 * 2 * 32 * 16); // after 8192 halves

  {
    long n4 = nES / 4;
    cvt_table<<<(int)((n4 + 255) / 256), 256, 0, stream>>>(es, tab, n4);
    int tot1 = KS1 * 16 * 32;   // 36864
    pack_weight<<<(tot1 + 255) / 256, 256, 0, stream>>>(Wp1, wp1p, KS1, 256);
    int tot0 = KS0 * 16 * 32;   // 1024
    pack_weight<<<(tot0 + 255) / 256, 256, 0, stream>>>(Wse2, wse2p, KS0, 256);
    int tot2 = KS2 * 16 * 32;   // 4096
    pack_weight<<<(tot2 + 255) / 256, 256, 0, stream>>>(Wp2, wp2p, KS2, 256);
    pack_heads<<<2, 256, 0, stream>>>(Wa, Ws, Wc, ba, bs, bc, whp, hb);
  }

  int nBlocks = (K + MROWS - 1) / MROWS;
  edge_main<<<nBlocks, 256, SMEM_BYTES, stream>>>(tab, pidx, oidx, corners,
                                                  Wse1, bse1, bse2,
                                                  bp1, lng, lnb, bp2,
                                                  wp1p, wse2p, wp2p, whp, hb,
                                                  out, K);
}
